// BiMamba_49306224558544
// MI455X (gfx1250) — matrix operand order repaired and checked by reading, compiled, not run on hardware
//
#include <hip/hip_runtime.h>
#include <hip/hip_bf16.h>

// ---------- types ----------
typedef __attribute__((ext_vector_type(16))) __bf16 v16bf;
typedef __attribute__((ext_vector_type(8)))  float  v8f;
typedef __attribute__((ext_vector_type(4)))  unsigned int vu4;
typedef unsigned short bf16_t;

union BFrag { vu4 u[2]; v16bf v; };

#define D_MODEL 256
#define D_INNER 512
#define D_STATE 16
#define DT_RANK 16
#define BATCH   4
#define SEQL    8192
#define MT      (BATCH * SEQL)   // 32768 rows

__device__ __forceinline__ bf16_t f2bf(float f) {
    unsigned u = __float_as_uint(f);
    unsigned r = u + 0x7FFFu + ((u >> 16) & 1u);   // RNE
    return (bf16_t)(r >> 16);
}
__device__ __forceinline__ float bf2f(bf16_t h) {
    return __uint_as_float(((unsigned)h) << 16);
}
__device__ __forceinline__ float silu(float x) {
    return x / (1.0f + __expf(-x));
}

// ---------- 1. transpose (B,256,L) f32 -> (B,L,256) bf16, fwd + reversed ----------
__global__ void transpose_in_kernel(const float* __restrict__ x,
                                    bf16_t* __restrict__ xin_f,
                                    bf16_t* __restrict__ xin_b) {
    __shared__ float tile[32][33];
    int tx = threadIdx.x;            // 0..31 (along L on read)
    int ty = threadIdx.y;            // 0..7
    int l0 = blockIdx.x * 32;
    int k0 = blockIdx.y * 32;
    int b  = blockIdx.z;
    #pragma unroll
    for (int i = 0; i < 4; ++i) {
        int k = k0 + ty + i * 8;
        tile[ty + i * 8][tx] = x[((size_t)b * D_MODEL + k) * SEQL + l0 + tx];
    }
    __syncthreads();
    #pragma unroll
    for (int i = 0; i < 4; ++i) {
        int l = l0 + ty + i * 8;
        bf16_t v = f2bf(tile[tx][ty + i * 8]);     // element (k0+tx, l)
        xin_f[((size_t)b * SEQL + l) * D_MODEL + k0 + tx] = v;
        xin_b[((size_t)b * SEQL + (SEQL - 1 - l)) * D_MODEL + k0 + tx] = v;
    }
}

// ---------- 2. f32 -> bf16 weight convert ----------
__global__ void cvt_bf16_kernel(const float* __restrict__ src, bf16_t* __restrict__ dst, int n) {
    int i = blockIdx.x * 256 + threadIdx.x;
    if (i < n) dst[i] = f2bf(src[i]);
}

// ---------- 3. fused output weights: Wc[o,k] (256 x 1024) bf16 ----------
__global__ void fuse_w_kernel(const float* __restrict__ proj_w,   // (256,512)
                              const float* __restrict__ fow,      // (256,512)
                              const float* __restrict__ bow,      // (256,512)
                              bf16_t* __restrict__ Wc) {
    int idx = blockIdx.x * 256 + threadIdx.x;        // 256*1024
    int o = idx >> 10;
    int k = idx & 1023;
    const float* ow = (k < 512) ? fow : bow;
    int d = k & 511;
    int coff = (k < 512) ? 0 : 256;
    float acc = 0.f;
    #pragma unroll 8
    for (int c = 0; c < 256; ++c)
        acc = fmaf(proj_w[(size_t)o * 512 + coff + c], ow[(size_t)c * 512 + d], acc);
    Wc[idx] = f2bf(acc);
}

// ---------- 4. generic WMMA bf16 GEMM: Out[M,N] = A[M,K] * W[N,K]^T ----------
template <int NTW, bool OUTBF>
__global__ void gemm_bf16_kernel(const bf16_t* __restrict__ A,
                                 const bf16_t* __restrict__ W,
                                 void* __restrict__ Out,
                                 int M, int N, int K) {
    int wg   = blockIdx.x * 8 + (threadIdx.x >> 5);
    int numWavesN = N / (16 * NTW);
    int mtile = wg / numWavesN;
    int ntb   = (wg % numWavesN) * NTW;
    if (mtile * 16 >= M) return;
    int lane = threadIdx.x & 31;
    int half = lane >> 4;
    int mr   = lane & 15;
    int mrow = mtile * 16 + mr;

    v8f acc[NTW];
    #pragma unroll
    for (int j = 0; j < NTW; ++j)
        #pragma unroll
        for (int e = 0; e < 8; ++e) acc[j][e] = 0.0f;

    const bf16_t* Ap = A + (size_t)mrow * K;
    for (int k0 = 0; k0 < K; k0 += 32) {
        BFrag a;
        a.u[0] = *(const vu4*)(Ap + k0 + half * 8);
        a.u[1] = *(const vu4*)(Ap + k0 + 16 + half * 8);
        #pragma unroll
        for (int j = 0; j < NTW; ++j) {
            int ncol = (ntb + j) * 16 + mr;
            const bf16_t* Wp = W + (size_t)ncol * K + k0 + half * 8;
            BFrag bfr;
            bfr.u[0] = *(const vu4*)(Wp);
            bfr.u[1] = *(const vu4*)(Wp + 16);
            acc[j] = __builtin_amdgcn_wmma_f32_16x16x32_bf16(
                false, a.v, false, bfr.v, (short)0, acc[j], false, false);
        }
    }
    #pragma unroll
    for (int j = 0; j < NTW; ++j) {
        int col = (ntb + j) * 16 + mr;
        #pragma unroll
        for (int v = 0; v < 8; ++v) {
            int row = mtile * 16 + v + 8 * half;
            if (OUTBF) ((bf16_t*)Out)[(size_t)row * N + col] = f2bf(acc[j][v]);
            else       ((float*)Out)[(size_t)row * N + col]  = acc[j][v];
        }
    }
}

// ---------- 5. depthwise causal conv (width 4) + SiLU ----------
__global__ void conv_silu_kernel(const bf16_t* __restrict__ xz,    // (MT,1024), xi = cols 0..511
                                 const float* __restrict__ conv_w, // (512,1,4)
                                 const float* __restrict__ conv_b,
                                 bf16_t* __restrict__ xc) {        // (MT,512)
    size_t tid = (size_t)blockIdx.x * 256 + threadIdx.x;
    int d = (int)(tid & 511);
    size_t m = tid >> 9;
    int l = (int)(m & (SEQL - 1));
    size_t mb = m - l;
    float acc = conv_b[d];
    #pragma unroll
    for (int j = 0; j < 4; ++j) {
        int lj = l - 3 + j;
        if (lj >= 0)
            acc = fmaf(conv_w[d * 4 + j], bf2f(xz[(mb + lj) * 1024 + d]), acc);
    }
    xc[m * 512 + d] = f2bf(silu(acc));
}

// ---------- 6. dt = softplus(dbc[:, :16] @ dtproj_w^T + b) ----------
__global__ void dt_kernel(const float* __restrict__ dbc,       // (MT,48)
                          const float* __restrict__ dtproj_w,  // (512,16)
                          const float* __restrict__ dtproj_b,
                          bf16_t* __restrict__ dt_out) {       // (MT,512)
    size_t tid = (size_t)blockIdx.x * 256 + threadIdx.x;
    int d = (int)(tid & 511);
    size_t m = tid >> 9;
    const float* row = dbc + m * 48;
    float acc = dtproj_b[d];
    #pragma unroll
    for (int r = 0; r < 16; ++r)
        acc = fmaf(row[r], dtproj_w[d * 16 + r], acc);
    float sp = (acc > 20.f) ? acc : log1pf(__expf(acc));
    dt_out[m * 512 + d] = f2bf(sp);
}

// ---------- 7. selective scan + skip + gate (both directions via blockIdx.y) ----------
__global__ void scan_kernel(const float*  __restrict__ dbc0, const bf16_t* __restrict__ dt0,
                            const bf16_t* __restrict__ xc0,  const bf16_t* __restrict__ xz0,
                            const float*  __restrict__ Al0,  const float*  __restrict__ Dk0,
                            bf16_t* __restrict__ y0,
                            const float*  __restrict__ dbc1, const bf16_t* __restrict__ dt1,
                            const bf16_t* __restrict__ xc1,  const bf16_t* __restrict__ xz1,
                            const float*  __restrict__ Al1,  const float*  __restrict__ Dk1,
                            bf16_t* __restrict__ y1) {
    int dir = blockIdx.y;
    const float*  dbc = dir ? dbc1 : dbc0;
    const bf16_t* dtb = dir ? dt1  : dt0;
    const bf16_t* xcb = dir ? xc1  : xc0;
    const bf16_t* xzb = dir ? xz1  : xz0;
    const float*  Al  = dir ? Al1  : Al0;
    const float*  Dk  = dir ? Dk1  : Dk0;
    bf16_t*       yb  = dir ? y1   : y0;

    int wave = threadIdx.x >> 5;
    int lane = threadIdx.x & 31;
    int n    = lane & 15;            // state index
    int dsub = lane >> 4;
    int bx   = blockIdx.x;           // 128 blocks: b = bx>>5, 16 channels each
    int b    = bx >> 5;
    int d    = (bx & 31) * 16 + wave * 2 + dsub;

    float Aval = -__expf(Al[d * 16 + n]);
    float Dv   = Dk[d];
    float h    = 0.0f;
    size_t mBase = (size_t)b * SEQL;

    for (int t = 0; t < SEQL; ++t) {
        size_t m = mBase + t;
        if (t + 8 < SEQL)
            __builtin_prefetch((const void*)(dbc + (m + 8) * 48), 0, 0);
        float Bv = dbc[m * 48 + 16 + n];
        float Cv = dbc[m * 48 + 32 + n];
        float dt = bf2f(dtb[m * 512 + d]);
        float u  = bf2f(xcb[m * 512 + d]);
        float dA = __expf(dt * Aval);
        h = fmaf(dA, h, dt * u * Bv);
        float p = h * Cv;
        p += __shfl_xor(p, 8, 32);
        p += __shfl_xor(p, 4, 32);
        p += __shfl_xor(p, 2, 32);
        p += __shfl_xor(p, 1, 32);
        if (n == 0) {
            float z = bf2f(xzb[m * 1024 + 512 + d]);
            float y = (p + u * Dv) * silu(z);
            yb[m * 512 + d] = f2bf(y);
        }
    }
}

// ---------- 8. final fused GEMM: out[b,o,l] = Wf*y_f + Wb*y_b + bias ----------
__global__ void gemm_final_kernel(const bf16_t* __restrict__ yf,
                                  const bf16_t* __restrict__ ybk,
                                  const bf16_t* __restrict__ Wc,   // (256,1024)
                                  const float*  __restrict__ proj_b,
                                  float* __restrict__ out) {       // (B,256,L)
    int wg    = blockIdx.x * 8 + (threadIdx.x >> 5);  // 8192 waves
    int mtile = wg >> 2;
    int ntb   = (wg & 3) * 4;
    int lane  = threadIdx.x & 31;
    int half  = lane >> 4;
    int mr    = lane & 15;
    int mrow  = mtile * 16 + mr;

    v8f acc[4];
    #pragma unroll
    for (int j = 0; j < 4; ++j)
        #pragma unroll
        for (int e = 0; e < 8; ++e) acc[j][e] = 0.0f;

    for (int k0 = 0; k0 < 1024; k0 += 32) {
        const bf16_t* Abase = (k0 < 512) ? yf : ybk;
        int kk = k0 & 511;
        BFrag a;
        a.u[0] = *(const vu4*)(Abase + (size_t)mrow * 512 + kk + half * 8);
        a.u[1] = *(const vu4*)(Abase + (size_t)mrow * 512 + kk + 16 + half * 8);
        #pragma unroll
        for (int j = 0; j < 4; ++j) {
            int ncol = (ntb + j) * 16 + mr;
            const bf16_t* Wp = Wc + (size_t)ncol * 1024 + k0 + half * 8;
            BFrag bfr;
            bfr.u[0] = *(const vu4*)(Wp);
            bfr.u[1] = *(const vu4*)(Wp + 16);
            acc[j] = __builtin_amdgcn_wmma_f32_16x16x32_bf16(
                false, a.v, false, bfr.v, (short)0, acc[j], false, false);
        }
    }
    #pragma unroll
    for (int j = 0; j < 4; ++j) {
        int col = (ntb + j) * 16 + mr;
        float bias = proj_b[col];
        #pragma unroll
        for (int v = 0; v < 8; ++v) {
            int m = mtile * 16 + v + 8 * half;
            int b = m >> 13;          // /8192
            int l = m & (SEQL - 1);
            out[((size_t)b * 256 + col) * SEQL + l] = acc[j][v] + bias;
        }
    }
}

// ---------- host orchestration ----------
extern "C" void kernel_launch(void* const* d_in, const int* in_sizes, int n_in,
                              void* d_out, int out_size, void* d_ws, size_t ws_size,
                              hipStream_t stream) {
    const float* x = (const float*)d_in[0];
    // per-direction params: [in_w, conv_w, conv_b, xproj_w, dtproj_w, dtproj_b, A_log, Dskip, out_w]
    const float* P[2][9];
    for (int i = 0; i < 9; ++i) { P[0][i] = (const float*)d_in[1 + i]; P[1][i] = (const float*)d_in[10 + i]; }
    const float* proj_w = (const float*)d_in[19];
    const float* proj_b = (const float*)d_in[20];
    float* out = (float*)d_out;

    // workspace carve-up (≈ 384 MB)
    char* base = (char*)d_ws;
    size_t off = 0;
    auto carve = [&](size_t bytes) -> char* {
        char* p = base + off;
        off = (off + bytes + 255) & ~(size_t)255;
        return p;
    };
    bf16_t* xin[2];  for (int d = 0; d < 2; ++d) xin[d] = (bf16_t*)carve((size_t)MT * 256 * 2);
    bf16_t* xz[2];   for (int d = 0; d < 2; ++d) xz[d]  = (bf16_t*)carve((size_t)MT * 1024 * 2);
    bf16_t* xc[2];   for (int d = 0; d < 2; ++d) xc[d]  = (bf16_t*)carve((size_t)MT * 512 * 2);
    float*  dbc[2];  for (int d = 0; d < 2; ++d) dbc[d] = (float*) carve((size_t)MT * 48 * 4);
    bf16_t* dtb[2];  for (int d = 0; d < 2; ++d) dtb[d] = (bf16_t*)carve((size_t)MT * 512 * 2);
    bf16_t* ybuf[2]; for (int d = 0; d < 2; ++d) ybuf[d]= (bf16_t*)carve((size_t)MT * 512 * 2);
    bf16_t* w_in[2]; for (int d = 0; d < 2; ++d) w_in[d]= (bf16_t*)carve((size_t)1024 * 256 * 2);
    bf16_t* w_xp[2]; for (int d = 0; d < 2; ++d) w_xp[d]= (bf16_t*)carve((size_t)48 * 512 * 2);
    bf16_t* Wc = (bf16_t*)carve((size_t)256 * 1024 * 2);

    // 1. input transpose + bf16 (fwd + reversed)
    transpose_in_kernel<<<dim3(SEQL / 32, D_MODEL / 32, BATCH), dim3(32, 8), 0, stream>>>(x, xin[0], xin[1]);

    // 2. weight conversions + fused output weight
    for (int d = 0; d < 2; ++d) {
        cvt_bf16_kernel<<<(1024 * 256 + 255) / 256, 256, 0, stream>>>(P[d][0], w_in[d], 1024 * 256);
        cvt_bf16_kernel<<<(48 * 512 + 255) / 256, 256, 0, stream>>>(P[d][3], w_xp[d], 48 * 512);
    }
    fuse_w_kernel<<<(256 * 1024) / 256, 256, 0, stream>>>(proj_w, P[0][8], P[1][8], Wc);

    // 3. per-direction pipeline up to scan inputs
    for (int d = 0; d < 2; ++d) {
        // in_proj: (MT,256)x(1024,256)^T -> (MT,1024) bf16; 32768 waves
        gemm_bf16_kernel<4, true><<<4096, 256, 0, stream>>>(xin[d], w_in[d], xz[d], MT, 1024, 256);
        // depthwise conv + silu
        conv_silu_kernel<<<(MT * 512) / 256, 256, 0, stream>>>(xz[d], P[d][1], P[d][2], xc[d]);
        // x_proj: (MT,512)x(48,512)^T -> (MT,48) f32; 2048 waves
        gemm_bf16_kernel<3, false><<<256, 256, 0, stream>>>(xc[d], w_xp[d], dbc[d], MT, 48, 512);
        // dt projection + softplus
        dt_kernel<<<(MT * 512) / 256, 256, 0, stream>>>(dbc[d], P[d][4], P[d][5], dtb[d]);
    }

    // 4. both scans concurrently (gridDim.y = direction)
    scan_kernel<<<dim3(128, 2), 256, 0, stream>>>(
        dbc[0], dtb[0], xc[0], xz[0], P[0][6], P[0][7], ybuf[0],
        dbc[1], dtb[1], xc[1], xz[1], P[1][6], P[1][7], ybuf[1]);

    // 5. fused out_proj + final projection, transposed store
    gemm_final_kernel<<<1024, 256, 0, stream>>>(ybuf[0], ybuf[1], Wc, proj_b, out);

    (void)in_sizes; (void)n_in; (void)out_size; (void)ws_size;
}
